// MyLSTM_18356690223257
// MI455X (gfx1250) — matrix operand order repaired and checked by reading, not compiled, not run on hardware
//
#include <hip/hip_runtime.h>
#include <hip/hip_bf16.h>
#include <stdint.h>

// ---------------------------------------------------------------------------
// LSTM forward for MI455X (gfx1250, wave32, WMMA 16x16x32 f16 -> f32 acc)
// B=32, S=1024, I=H=512
//
// Phase 1: f32 -> f16 conversion of x and all 9 weight matrices.
// Phase 2: input projections (WMMA GEMM), gate pre-activations written
//          PRE-SWIZZLED in WMMA C/D fragment order. A-panel staged to LDS
//          with gfx1250 async copies (global_load_async_to_lds_b128).
// Phase 3: sequential recurrence, 2 workgroups x 16 batch rows (M=16 tile),
//          h in LDS (A-fragment order), c in registers. Software-pipelined
//          B-fragment double buffering + sched_barrier fences so each load
//          group's L2 latency hides behind a group of 8 WMMAs.
// Phase 4: output projection  out = hs[:,512:,:] @ Wl^T + bl.
// ---------------------------------------------------------------------------

typedef __attribute__((ext_vector_type(16))) _Float16 v16h;
typedef __attribute__((ext_vector_type(8)))  float    v8f;

#define BATCH 32
#define SEQ   1024
#define DIM   512       // I == H == 512
#define KT    16        // K tiles of 32 along DIM
#define WELEM (DIM * DIM)

__device__ __forceinline__ float sigmoidf_(float x) {
    return 1.0f / (1.0f + __expf(-x));
}

// VALU-only tanh (clamped so exp stays finite)
__device__ __forceinline__ float tanhf_(float x) {
    x = fminf(fmaxf(x, -15.0f), 15.0f);
    float e = __expf(2.0f * x);
    return (e - 1.0f) / (e + 1.0f);
}

__device__ __forceinline__ v8f wmma16x16x32(v16h a, v16h b, v8f c) {
    // D = A(16x32 f16) * B(32x16 f16) + C(16x16 f32)
    return __builtin_amdgcn_wmma_f32_16x16x32_f16(
        /*neg_a=*/false, a, /*neg_b=*/false, b,
        /*c_mod=*/(short)0, c, /*reuse_a=*/false, /*reuse_b=*/false);
}

// A-fragment (16x32 f16) from LDS stored as 1024 16-byte chunks:
// chunk(kt, lane', hc) holds row m=lane'&15, k = kt*32 + (lane'<16?0:8) + hc*16 .. +7
__device__ __forceinline__ v16h ldsAfrag(const uint4* ldsA, int kt, int lane) {
    union { uint4 u[2]; v16h v; } x;
    const uint4* p = ldsA + (kt * 64 + lane * 2);
    x.u[0] = p[0];
    x.u[1] = p[1];
    return x.v;
}

// B-fragment (32x16 f16): lane -> column n (caller passes W + n*512);
// lanes 0-15 hold K=kt*32+{0..7,16..23}, lanes 16-31 hold +{8..15,24..31} -> two 16B chunks.
__device__ __forceinline__ v16h gmemBfrag(const _Float16* wrow, int kt, int lane) {
    union { uint4 u[2]; v16h v; } x;
    const uint4* p = reinterpret_cast<const uint4*>(wrow + kt * 32 + ((lane & 16) ? 8 : 0));
    x.u[0] = p[0];
    x.u[1] = p[2];
    return x.v;
}

// Cooperatively stage a 16x512 f16 A-panel into LDS A-fragment chunk order
// using gfx1250 async copies (tracked by ASYNCcnt, no VGPR data movement).
__device__ __forceinline__ void stageApanelAsync(uint4* ldsA, const _Float16* src,
                                                 size_t rowStride, int tid, int nthreads) {
    for (int c = tid; c < 1024; c += nthreads) {
        int kt   = c >> 6;
        int rem  = c & 63;
        int lp   = rem >> 1;          // lane' 0..31
        int hc   = rem & 1;
        int m    = lp & 15;
        int kofs = kt * 32 + ((lp & 16) ? 8 : 0) + hc * 16;
        const void* gaddr = (const void*)(src + (size_t)m * rowStride + kofs);
        unsigned    laddr = (unsigned)(uintptr_t)(ldsA + c);   // LDS byte offset
        asm volatile("global_load_async_to_lds_b128 %0, %1, off"
                     :: "v"(laddr), "v"(gaddr) : "memory");
    }
    asm volatile("s_wait_asynccnt 0x0" ::: "memory");
}

// ---------------------------------------------------------------------------
// Phase 1: f32 -> f16
// ---------------------------------------------------------------------------
__global__ void cvt_f32_to_f16(const float* __restrict__ in, _Float16* __restrict__ out, int n) {
    int i = blockIdx.x * blockDim.x + threadIdx.x;
    if (i < n) out[i] = (_Float16)in[i];
}

// Software-pipelined K-loop over 8 output tiles: double-buffered B fragments,
// sched_barrier(0) fences so loads group ahead of each 8-WMMA burst.
__device__ __forceinline__ void gemmK8(const uint4* ldsA, const _Float16* const (&row)[8],
                                       int lane, v8f (&acc)[8]) {
    v16h bb0[8], bb1[8];
#pragma unroll
    for (int u = 0; u < 8; ++u) bb0[u] = gmemBfrag(row[u], 0, lane);
#pragma unroll
    for (int kt = 0; kt < KT; kt += 2) {
        v16h a0 = ldsAfrag(ldsA, kt, lane);
#pragma unroll
        for (int u = 0; u < 8; ++u) bb1[u] = gmemBfrag(row[u], kt + 1, lane);
        __builtin_amdgcn_sched_barrier(0);
#pragma unroll
        for (int u = 0; u < 8; ++u) acc[u] = wmma16x16x32(a0, bb0[u], acc[u]);

        v16h a1 = ldsAfrag(ldsA, kt + 1, lane);
        if (kt + 2 < KT) {
#pragma unroll
            for (int u = 0; u < 8; ++u) bb0[u] = gmemBfrag(row[u], kt + 2, lane);
        }
        __builtin_amdgcn_sched_barrier(0);
#pragma unroll
        for (int u = 0; u < 8; ++u) acc[u] = wmma16x16x32(a1, bb1[u], acc[u]);
    }
}

// ---------------------------------------------------------------------------
// Phase 2: input projections. grid = (SEQ, 2), block = 512 (16 waves).
// Wave w computes 4 gates x 2 N-tiles (u = g*2+j, nt = w*2+j).
// gate_buf index: ((((t*4+g)*2+half)*32+nt)*32+lane)*8 + r
// ---------------------------------------------------------------------------
__global__ __launch_bounds__(512)
void lstm_input_proj(const _Float16* __restrict__ xh,
                     const _Float16* __restrict__ Ww,   // [Wz,Wi,Wf,Wo] f16
                     const float* __restrict__ bz, const float* __restrict__ bi,
                     const float* __restrict__ bf, const float* __restrict__ bo,
                     float* __restrict__ gate_buf) {
    __shared__ uint4 ldsA[1024];
    const int t    = blockIdx.x;
    const int half = blockIdx.y;
    const int tid  = threadIdx.x;
    const int lane = tid & 31;
    const int w    = tid >> 5;

    stageApanelAsync(ldsA, xh + ((size_t)(half * 16) * SEQ + (size_t)t) * DIM,
                     (size_t)SEQ * DIM, tid, 512);
    __syncthreads();

    const float* bptr[4] = { bz, bi, bf, bo };
    int ncol[2];
#pragma unroll
    for (int j = 0; j < 2; ++j) ncol[j] = (w * 2 + j) * 16 + (lane & 15);

    const _Float16* wrow[8];
#pragma unroll
    for (int u = 0; u < 8; ++u) {
        int g = u >> 1, j = u & 1;
        wrow[u] = Ww + (size_t)g * WELEM + (size_t)ncol[j] * DIM;
    }

    v8f acc[8];
#pragma unroll
    for (int u = 0; u < 8; ++u) {
        float bv = bptr[u >> 1][ncol[u & 1]];
#pragma unroll
        for (int r = 0; r < 8; ++r) acc[u][r] = bv;
    }

    gemmK8(ldsA, wrow, lane, acc);

#pragma unroll
    for (int u = 0; u < 8; ++u) {
        int g = u >> 1, j = u & 1;
        int nt = w * 2 + j;
        size_t base = (((((size_t)t * 4 + g) * 2 + half) * 32 + nt) * 32 + lane) * 8;
        float4* dst = reinterpret_cast<float4*>(gate_buf + base);
        dst[0] = make_float4(acc[u][0], acc[u][1], acc[u][2], acc[u][3]);
        dst[1] = make_float4(acc[u][4], acc[u][5], acc[u][6], acc[u][7]);
    }
}

// ---------------------------------------------------------------------------
// Phase 3: recurrence. grid = 2, block = 512 (16 waves).
// Wave w owns N-tiles nt = w*2, w*2+1 for all 4 gates; acc order u = g*2+j.
// ---------------------------------------------------------------------------
__global__ __launch_bounds__(512)
void lstm_recurrence(const _Float16* __restrict__ Rw,  // [Rz,Ri,Rf,Ro] f16
                     const float* __restrict__ gate_buf,
                     const float* __restrict__ brz, const float* __restrict__ bri,
                     const float* __restrict__ brf, const float* __restrict__ bro,
                     _Float16* __restrict__ hs) {
    __shared__ uint4 ldsA[1024];
    const int half = blockIdx.x;
    const int tid  = threadIdx.x;
    const int lane = tid & 31;
    const int w    = tid >> 5;

    // h0 = 0
    uint4 zz; zz.x = zz.y = zz.z = zz.w = 0u;
    for (int c = tid; c < 1024; c += 512) ldsA[c] = zz;
    __syncthreads();

    const float* brp[4] = { brz, bri, brf, bro };
    int ncol[2];
#pragma unroll
    for (int j = 0; j < 2; ++j) ncol[j] = (w * 2 + j) * 16 + (lane & 15);

    float bias[8];
    const _Float16* rrow[8];
#pragma unroll
    for (int u = 0; u < 8; ++u) {
        int g = u >> 1, j = u & 1;
        bias[u] = brp[g][ncol[j]];
        rrow[u] = Rw + (size_t)g * WELEM + (size_t)ncol[j] * DIM;
    }

    v8f cst[2];
#pragma unroll
    for (int j = 0; j < 2; ++j)
#pragma unroll
        for (int r = 0; r < 8; ++r) cst[j][r] = 0.0f;

    _Float16* ldsH = reinterpret_cast<_Float16*>(ldsA);

    for (int t = 0; t < SEQ; ++t) {
        // init accumulators with precomputed x-projection + recurrent bias
        v8f acc[8];
#pragma unroll
        for (int u = 0; u < 8; ++u) {
            int g = u >> 1, j = u & 1;
            int nt = w * 2 + j;
            size_t base = (((((size_t)t * 4 + g) * 2 + half) * 32 + nt) * 32 + lane) * 8;
            const float4* gp = reinterpret_cast<const float4*>(gate_buf + base);
            float4 g0 = gp[0], g1 = gp[1];
            float bv = bias[u];
            acc[u][0] = g0.x + bv; acc[u][1] = g0.y + bv;
            acc[u][2] = g0.z + bv; acc[u][3] = g0.w + bv;
            acc[u][4] = g1.x + bv; acc[u][5] = g1.y + bv;
            acc[u][6] = g1.z + bv; acc[u][7] = g1.w + bv;
        }

        // prefetch next step's gate tiles into cache while we do the GEMM
        if (t + 1 < SEQ) {
#pragma unroll
            for (int u = 0; u < 8; ++u) {
                int g = u >> 1, j = u & 1;
                int nt = w * 2 + j;
                size_t base = (((((size_t)(t + 1) * 4 + g) * 2 + half) * 32 + nt) * 32 + lane) * 8;
                __builtin_prefetch(gate_buf + base, 0, 3);
            }
        }

        // recurrent GEMM: acc += h(16x512) @ Rg^T tile (pipelined WMMA bursts)
        gemmK8(ldsA, rrow, lane, acc);

        __syncthreads();   // all waves done reading h(t-1) from LDS

        // elementwise gate math; scatter h(t) back into LDS A-fragment layout
#pragma unroll
        for (int j = 0; j < 2; ++j) {
            const int k  = ncol[j];
            const int kt = k >> 5;
            const int ko = k & 31;
            const int hc = (ko >> 4) & 1;
            const int rm = ko & 15;
#pragma unroll
            for (int r = 0; r < 8; ++r) {
                float z = tanhf_(acc[0 + j][r]);        // u = g*2+j
                float i = sigmoidf_(acc[2 + j][r]);
                float f = sigmoidf_(acc[4 + j][r]);
                float o = sigmoidf_(acc[6 + j][r]);
                float cv = z * i + f * cst[j][r];
                cst[j][r] = cv;
                float hv = o * tanhf_(cv);
                _Float16 hh = (_Float16)hv;

                int m  = r + ((lane >> 4) << 3);        // row in M-tile
                int lp = (rm < 8) ? m : (m + 16);
                int off = rm & 7;
                ldsH[(kt * 64 + lp * 2 + hc) * 8 + off] = hh;

                if (t >= SEQ / 2) {
                    int b = half * 16 + m;
                    hs[((size_t)b * (SEQ / 2) + (size_t)(t - SEQ / 2)) * DIM + k] = hh;
                }
            }
        }
        __syncthreads();   // h(t) fully written before next step reads it
    }
}

// ---------------------------------------------------------------------------
// Phase 4: out[b, t', :] = hs[b, t', :] @ Wl^T + bl.
// Rows (b,t') tiled by 16; grid = 1024 blocks, 256 threads (8 waves x 4 tiles).
// ---------------------------------------------------------------------------
__global__ __launch_bounds__(256)
void lstm_out_proj(const _Float16* __restrict__ hsh,
                   const _Float16* __restrict__ Wlh,
                   const float* __restrict__ bl,
                   float* __restrict__ out) {
    __shared__ uint4 ldsA[1024];
    const int mt  = blockIdx.x;
    const int b   = mt >> 5;             // 512/16 = 32 row-tiles per batch entry
    const int t0  = (mt & 31) << 4;
    const int tid = threadIdx.x;
    const int lane = tid & 31;
    const int w    = tid >> 5;

    stageApanelAsync(ldsA, hsh + ((size_t)b * (SEQ / 2) + t0) * DIM, (size_t)DIM, tid, 256);
    __syncthreads();

    int ncol[4];
    const _Float16* wrow[4];
#pragma unroll
    for (int j = 0; j < 4; ++j) {
        ncol[j] = (w * 4 + j) * 16 + (lane & 15);
        wrow[j] = Wlh + (size_t)ncol[j] * DIM;
    }

    v8f acc[4];
#pragma unroll
    for (int j = 0; j < 4; ++j) {
        float bv = bl[ncol[j]];
#pragma unroll
        for (int r = 0; r < 8; ++r) acc[j][r] = bv;
    }

    v16h bb0[4], bb1[4];
#pragma unroll
    for (int j = 0; j < 4; ++j) bb0[j] = gmemBfrag(wrow[j], 0, lane);
#pragma unroll
    for (int kt = 0; kt < KT; kt += 2) {
        v16h a0 = ldsAfrag(ldsA, kt, lane);
#pragma unroll
        for (int j = 0; j < 4; ++j) bb1[j] = gmemBfrag(wrow[j], kt + 1, lane);
        __builtin_amdgcn_sched_barrier(0);
#pragma unroll
        for (int j = 0; j < 4; ++j) acc[j] = wmma16x16x32(a0, bb0[j], acc[j]);

        v16h a1 = ldsAfrag(ldsA, kt + 1, lane);
        if (kt + 2 < KT) {
#pragma unroll
            for (int j = 0; j < 4; ++j) bb0[j] = gmemBfrag(wrow[j], kt + 2, lane);
        }
        __builtin_amdgcn_sched_barrier(0);
#pragma unroll
        for (int j = 0; j < 4; ++j) acc[j] = wmma16x16x32(a1, bb1[j], acc[j]);
    }

#pragma unroll
    for (int j = 0; j < 4; ++j)
#pragma unroll
        for (int r = 0; r < 8; ++r) {
            int m = r + ((lane >> 4) << 3);
            out[((size_t)b * (SEQ / 2) + (size_t)(t0 + m)) * DIM + ncol[j]] = acc[j][r];
        }
}

// ---------------------------------------------------------------------------
// Launch
// ---------------------------------------------------------------------------
extern "C" void kernel_launch(void* const* d_in, const int* in_sizes, int n_in,
                              void* d_out, int out_size, void* d_ws, size_t ws_size,
                              hipStream_t stream) {
    (void)in_sizes; (void)n_in; (void)out_size; (void)ws_size;

    const float* x   = (const float*)d_in[0];
    const int widx[9] = { 1, 3, 5, 7, 9, 11, 13, 15, 17 };  // Wz,Wi,Wf,Wo,Rz,Ri,Rf,Ro,Wl
    const float* bz  = (const float*)d_in[2];
    const float* bi  = (const float*)d_in[4];
    const float* bf  = (const float*)d_in[6];
    const float* bo  = (const float*)d_in[8];
    const float* brz = (const float*)d_in[10];
    const float* bri = (const float*)d_in[12];
    const float* brf = (const float*)d_in[14];
    const float* bro = (const float*)d_in[16];
    const float* bl  = (const float*)d_in[18];

    char* ws = (char*)d_ws;
    const size_t NX = (size_t)BATCH * SEQ * DIM;          // 16,777,216
    size_t off = 0;
    _Float16* xh = (_Float16*)(ws + off);   off += NX * sizeof(_Float16);                 // 32 MB
    _Float16* Wh = (_Float16*)(ws + off);   off += (size_t)9 * WELEM * sizeof(_Float16);  // 4.5 MB
    float* gate_buf = (float*)(ws + off);   off += (size_t)4 * SEQ * BATCH * DIM * sizeof(float); // 256 MB
    _Float16* hsbuf = (_Float16*)(ws + off); // 16 MB

    // Phase 1: conversions
    cvt_f32_to_f16<<<(int)((NX + 255) / 256), 256, 0, stream>>>(x, xh, (int)NX);
    for (int i = 0; i < 9; ++i) {
        cvt_f32_to_f16<<<(WELEM + 255) / 256, 256, 0, stream>>>(
            (const float*)d_in[widx[i]], Wh + (size_t)i * WELEM, WELEM);
    }

    // Phase 2: input projections (fragment-swizzled gate pre-activations)
    lstm_input_proj<<<dim3(SEQ, 2), 512, 0, stream>>>(
        xh, Wh, bz, bi, bf, bo, gate_buf);

    // Phase 3: sequential recurrence (2 independent batch-half chains)
    lstm_recurrence<<<2, 512, 0, stream>>>(
        Wh + (size_t)4 * WELEM, gate_buf, brz, bri, brf, bro, hsbuf);

    // Phase 4: output projection on second half of the sequence
    lstm_out_proj<<<1024, 256, 0, stream>>>(
        hsbuf, Wh + (size_t)8 * WELEM, bl, (float*)d_out);
}
